// SpatioTemporalAttention_78460462564113
// MI455X (gfx1250) — matrix operand order repaired and checked by reading, not compiled, not run on hardware
//
#include <hip/hip_runtime.h>

// ---------------------------------------------------------------------------
// MI455X (gfx1250) spatio-temporal attention, bf16 WMMA pipeline.
// B=2 T=32 C=1024 D=512, 8 heads x 64.  ~420 GFLOP -> compute bound ->
// everything matrix-shaped goes through v_wmma_f32_16x16x32_bf16.
// GEMM waves now compute 32x64 tiles (B-fragment reuse across two A m-tiles)
// to double WMMA per loaded byte and give the scheduler two independent
// accumulation chains.
// ---------------------------------------------------------------------------

typedef __bf16 bf16;
typedef __attribute__((ext_vector_type(16))) __bf16 v16bf;
typedef __attribute__((ext_vector_type(8)))  float  v8f;

union Frag { v16bf v; uint4 u[2]; };

// A-fragment (16x32, 16-bit): lane L<16 -> row L, K hk..hk+15,
// hk = 0 for lanes 0-15, 16 for lanes 16-31.  One contiguous 32B run.
__device__ inline v16bf load_a_frag(const bf16* p, size_t ld) {
  int lane = threadIdx.x & 31;
  const bf16* q = p + (size_t)(lane & 15) * ld + (size_t)((lane >> 4) << 4);
  Frag f;
  f.u[0] = *(const uint4*)(q);
  f.u[1] = *(const uint4*)(q + 8);
  return f.v;
}

// B-fragment (32x16, 16-bit) from a transposed [N][K] layout: lane n holds
// column n, K 0..15 (lanes 0-15) or 16..31 (lanes 16-31), contiguous 32B.
__device__ inline v16bf load_b_frag(const bf16* p, size_t ld) {
  int lane = threadIdx.x & 31;
  const bf16* q = p + (size_t)(lane & 15) * ld + (size_t)((lane >> 4) << 4);
  Frag f;
  f.u[0] = *(const uint4*)(q);
  f.u[1] = *(const uint4*)(q + 8);
  return f.v;
}

__device__ inline v8f wmma_bf16(v16bf a, v16bf b, v8f c) {
  return __builtin_amdgcn_wmma_f32_16x16x32_bf16(false, a, false, b, (short)0, c,
                                                 false, false);
}

// ---------------------------------------------------------------------------
// Weight transpose + fp32 -> bf16 cast:  wt[n*K + k] = w[k*N + n]
// ---------------------------------------------------------------------------
__global__ void tcast_kernel(const float* __restrict__ w, bf16* __restrict__ wt,
                             int K, int N) {
  int i = blockIdx.x * 256 + threadIdx.x;
  if (i >= K * N) return;
  int k = i / N, n = i - k * N;
  wt[(size_t)n * K + k] = (bf16)w[i];
}

// ---------------------------------------------------------------------------
// LayerNorm over last dim (512) + cast to bf16.  One wave32 per row.
// ---------------------------------------------------------------------------
__global__ __launch_bounds__(256)
void ln_cast_kernel(const float* __restrict__ x, const float* __restrict__ g,
                    const float* __restrict__ be, bf16* __restrict__ out,
                    int nrows) {
  int gwave = (blockIdx.x * 256 + threadIdx.x) >> 5;
  int lane  = threadIdx.x & 31;
  if (gwave >= nrows) return;
  const float4* row = (const float4*)(x + (size_t)gwave * 512);
  float4 v[4];
  float s = 0.f, s2 = 0.f;
#pragma unroll
  for (int i = 0; i < 4; ++i) {
    v[i] = row[lane + i * 32];
    s  += v[i].x + v[i].y + v[i].z + v[i].w;
    s2 += v[i].x * v[i].x + v[i].y * v[i].y + v[i].z * v[i].z + v[i].w * v[i].w;
  }
#pragma unroll
  for (int off = 16; off > 0; off >>= 1) {
    s  += __shfl_xor(s,  off, 32);
    s2 += __shfl_xor(s2, off, 32);
  }
  float mean = s * (1.f / 512.f);
  float var  = s2 * (1.f / 512.f) - mean * mean;
  float rstd = rsqrtf(var + 1e-5f);
  bf16* orow = out + (size_t)gwave * 512;
#pragma unroll
  for (int i = 0; i < 4; ++i) {
    int c = (lane + i * 32) * 4;
    orow[c + 0] = (bf16)((v[i].x - mean) * rstd * g[c + 0] + be[c + 0]);
    orow[c + 1] = (bf16)((v[i].y - mean) * rstd * g[c + 1] + be[c + 1]);
    orow[c + 2] = (bf16)((v[i].z - mean) * rstd * g[c + 2] + be[c + 2]);
    orow[c + 3] = (bf16)((v[i].w - mean) * rstd * g[c + 3] + be[c + 3]);
  }
}

// ---------------------------------------------------------------------------
// GEMM:  C[M,N] = A[M,K=512] * B[K,N], B given transposed as Bt[N][K] (bf16).
// One wave computes a 32x64 tile: per K-step, 2 A-frags + 4 B-frags feed
// 8 WMMAs (B reused across both m-tiles).  A/weights live in L2 (h = 64MB,
// W <= 1.5MB, both < 192MB L2), so no LDS staging is needed.
// Output: bf16 (no bias) or fp32 (+bias).
// ---------------------------------------------------------------------------
__global__ __launch_bounds__(256)
void gemm_bf16_kernel(const bf16* __restrict__ A, const bf16* __restrict__ Bt,
                      int M, int N, bf16* __restrict__ outb,
                      float* __restrict__ outf, const float* __restrict__ bias) {
  const int K = 512;
  int gwave  = (blockIdx.x * 256 + threadIdx.x) >> 5;
  int ntiles = N >> 6;
  int mt = gwave / ntiles;
  int nt = gwave - mt * ntiles;
  if (mt * 32 >= M) return;
  int lane = threadIdx.x & 31;
  const bf16* arow = A  + (size_t)mt * 32 * K;
  const bf16* bcol = Bt + (size_t)nt * 64 * K;
  v8f acc[2][4] = {};
  for (int k0 = 0; k0 < K; k0 += 32) {
    __builtin_prefetch(arow + (size_t)(lane & 15) * K + k0 + 128, 0, 1);
    __builtin_prefetch(arow + (size_t)(16 + (lane & 15)) * K + k0 + 128, 0, 1);
    v16bf a0 = load_a_frag(arow + k0, K);
    v16bf a1 = load_a_frag(arow + (size_t)16 * K + k0, K);
#pragma unroll
    for (int j = 0; j < 4; ++j) {
      v16bf b = load_b_frag(bcol + (size_t)j * 16 * K + k0, K);
      acc[0][j] = wmma_bf16(a0, b, acc[0][j]);
      acc[1][j] = wmma_bf16(a1, b, acc[1][j]);
    }
  }
  // D layout: VGPR r -> rows (r, r+8) over lane halves, N = lane&15.
  int n0 = lane & 15;
#pragma unroll
  for (int mi = 0; mi < 2; ++mi) {
    int mrow = mt * 32 + mi * 16 + ((lane >> 4) << 3);
#pragma unroll
    for (int j = 0; j < 4; ++j) {
      int col  = nt * 64 + j * 16 + n0;
      float bv = bias ? bias[col] : 0.f;
#pragma unroll
      for (int r = 0; r < 8; ++r) {
        size_t idx = (size_t)(mrow + r) * N + col;
        float val  = acc[mi][j][r] + bv;
        if (outf) outf[idx] = val;
        else      outb[idx] = (bf16)val;
      }
    }
  }
}

// ---------------------------------------------------------------------------
// Flash attention over strided token rows of the qkv buffer [tok][1536]
// (q: 0..511, k: 512..1023, v: 1024..1535, head h at h*64).
// Block = 8 waves = 8 heads; K tile [32x512] + V^T tile [512x32] staged in LDS
// (shared by all heads).  Per wave: Q held in registers (A-frags), S = Q.K^T
// via WMMA, online softmax using the wave32 D-layout (VGPR r = rows r/r+8 on
// lane halves -> width-16 shuffle reductions), P -> per-wave LDS tile ->
// A-frag (in-wave LDS ops are in-order, no barrier), O += P.V via WMMA.
//   TEMPORAL: 32 queries/wave (MT=2), keys = 32 (1 tile), row stride C.
//   SPATIAL : 16 queries/wave (MT=1), keys = 1024 (32 tiles), row stride 1.
// ---------------------------------------------------------------------------
template <int MT, bool TEMPORAL>
__global__ __launch_bounds__(256)
void flash_kernel(const bf16* __restrict__ qkv, bf16* __restrict__ attn_out) {
  const int T = 32, C = 1024, ROW = 1536;
  const float SCALE = 0.125f;  // 64^-0.5
  __shared__ __align__(16) bf16 Kls[32 * 512];   // [key][dim 0..511]
  __shared__ __align__(16) bf16 Vls[512 * 32];   // [dim][key]  (transposed)
  __shared__ __align__(16) bf16 Pls[8][MT * 16 * 32];
  int wave = threadIdx.x >> 5;
  int lane = threadIdx.x & 31;
  int head = wave;
  size_t qbase, kbase;
  int stride, nkeys;
  if (TEMPORAL) {
    int b = blockIdx.x / C, c = blockIdx.x - b * C;
    qbase = (size_t)b * T * C + c;
    kbase = qbase;
    stride = C;
    nkeys = 32;
  } else {
    int bt = blockIdx.x >> 6, qc = blockIdx.x & 63;
    qbase = (size_t)bt * C + qc * 16;
    kbase = (size_t)bt * C;
    stride = 1;
    nkeys = C;
  }

  // Q fragments, kept in registers for the whole key loop.
  v16bf qf[MT][2];
#pragma unroll
  for (int mt = 0; mt < MT; ++mt)
#pragma unroll
    for (int ks = 0; ks < 2; ++ks)
      qf[mt][ks] = load_a_frag(
          qkv + (qbase + (size_t)mt * 16 * stride) * ROW + head * 64 + ks * 32,
          (size_t)stride * ROW);

  v8f o[MT][4] = {};
  float mrow[MT][8], lrow[MT][8];
#pragma unroll
  for (int mt = 0; mt < MT; ++mt)
#pragma unroll
    for (int r = 0; r < 8; ++r) { mrow[mt][r] = -1e30f; lrow[mt][r] = 0.f; }

  int half = lane >> 4, n0 = lane & 15;

  for (int kt = 0; kt < nkeys / 32; ++kt) {
    // --- cooperative staging: K tile row-major, V tile transposed ---
    {
      int j   = threadIdx.x >> 3;          // key 0..31
      int off = (threadIdx.x & 7) * 64;    // dim chunk
      const bf16* src = qkv + (kbase + (size_t)(kt * 32 + j) * stride) * ROW;
      const uint4* ksrc = (const uint4*)(src + 512 + off);
      uint4* kdst = (uint4*)(&Kls[j * 512 + off]);
#pragma unroll
      for (int u = 0; u < 4; ++u) kdst[u] = ksrc[u];
      const bf16* vsrc = src + 1024 + off;
#pragma unroll
      for (int d = 0; d < 64; ++d) Vls[(off + d) * 32 + j] = vsrc[d];
    }
    __syncthreads();

#pragma unroll
    for (int mt = 0; mt < MT; ++mt) {
      // S[16 x 32] = Q . K^T   (two 16-key n-tiles)
      v8f s0 = {}, s1 = {};
#pragma unroll
      for (int ks = 0; ks < 2; ++ks) {
        v16bf b0 = load_b_frag(&Kls[0 * 16 * 512 + head * 64 + ks * 32], 512);
        v16bf b1 = load_b_frag(&Kls[1 * 16 * 512 + head * 64 + ks * 32], 512);
        s0 = wmma_bf16(qf[mt][ks], b0, s0);
        s1 = wmma_bf16(qf[mt][ks], b1, s1);
      }
      // --- online softmax per row (rows r / r+8 live on lane halves) ---
      bf16* prow_base = &Pls[wave][mt * 16 * 32];
#pragma unroll
      for (int r = 0; r < 8; ++r) {
        float a0 = s0[r] * SCALE, a1 = s1[r] * SCALE;
        float mx = fmaxf(a0, a1);
#pragma unroll
        for (int off = 8; off > 0; off >>= 1)
          mx = fmaxf(mx, __shfl_xor(mx, off, 16));
        float nm = fmaxf(mrow[mt][r], mx);
        float p0 = __expf(a0 - nm), p1 = __expf(a1 - nm);
        float ts = p0 + p1;
#pragma unroll
        for (int off = 8; off > 0; off >>= 1) ts += __shfl_xor(ts, off, 16);
        float alpha = __expf(mrow[mt][r] - nm);
        lrow[mt][r] = lrow[mt][r] * alpha + ts;
        mrow[mt][r] = nm;
#pragma unroll
        for (int j = 0; j < 4; ++j) o[mt][j][r] = o[mt][j][r] * alpha;
        bf16* pr = prow_base + (r + half * 8) * 32;
        pr[n0]      = (bf16)p0;
        pr[16 + n0] = (bf16)p1;
      }
      // --- O += P . V ---
      v16bf pf = load_a_frag(prow_base, 32);
#pragma unroll
      for (int j = 0; j < 4; ++j) {
        v16bf vf = load_b_frag(&Vls[(head * 64 + j * 16) * 32], 32);
        o[mt][j] = wmma_bf16(pf, vf, o[mt][j]);
      }
    }
    __syncthreads();
  }

  // --- epilogue: O / l -> bf16, out layout [tok][512] ---
#pragma unroll
  for (int mt = 0; mt < MT; ++mt)
#pragma unroll
    for (int r = 0; r < 8; ++r) {
      size_t qrow = qbase + (size_t)(mt * 16 + r + half * 8) * stride;
      float inv   = 1.f / lrow[mt][r];
#pragma unroll
      for (int j = 0; j < 4; ++j)
        attn_out[qrow * 512 + head * 64 + j * 16 + n0] =
            (bf16)(o[mt][j][r] * inv);
    }
}

// ---------------------------------------------------------------------------
extern "C" void kernel_launch(void* const* d_in, const int* in_sizes, int n_in,
                              void* d_out, int out_size, void* d_ws,
                              size_t ws_size, hipStream_t stream) {
  (void)in_sizes; (void)n_in; (void)out_size; (void)ws_size;
  const float* x       = (const float*)d_in[0];
  const float* gamma_t = (const float*)d_in[1];
  const float* beta_t  = (const float*)d_in[2];
  const float* w_qkv_t = (const float*)d_in[3];
  const float* w_out_t = (const float*)d_in[4];
  const float* b_out_t = (const float*)d_in[5];
  const float* gamma_s = (const float*)d_in[6];
  const float* beta_s  = (const float*)d_in[7];
  const float* w_qkv_s = (const float*)d_in[8];
  const float* w_out_s = (const float*)d_in[9];
  const float* b_out_s = (const float*)d_in[10];
  float* out = (float*)d_out;

  const int NTOK = 65536;   // B*T*C
  const int D = 512, NI3 = 1536;

  char* p = (char*)d_ws;
  auto take = [&](size_t bytes) {
    char* r = p;
    p += (bytes + 255) & ~(size_t)255;
    return r;
  };
  bf16*  wqT_t = (bf16*)take((size_t)NI3 * D * 2);
  bf16*  woT_t = (bf16*)take((size_t)D * D * 2);
  bf16*  wqT_s = (bf16*)take((size_t)NI3 * D * 2);
  bf16*  woT_s = (bf16*)take((size_t)D * D * 2);
  bf16*  h     = (bf16*)take((size_t)NTOK * D * 2);
  bf16*  qkv   = (bf16*)take((size_t)NTOK * NI3 * 2);
  bf16*  attn  = (bf16*)take((size_t)NTOK * D * 2);
  float* mid   = (float*)take((size_t)NTOK * D * 4);

  // Weights: transpose to [N][K] bf16 so WMMA B-frags are contiguous.
  tcast_kernel<<<(D * NI3 + 255) / 256, 256, 0, stream>>>(w_qkv_t, wqT_t, D, NI3);
  tcast_kernel<<<(D * D   + 255) / 256, 256, 0, stream>>>(w_out_t, woT_t, D, D);
  tcast_kernel<<<(D * NI3 + 255) / 256, 256, 0, stream>>>(w_qkv_s, wqT_s, D, NI3);
  tcast_kernel<<<(D * D   + 255) / 256, 256, 0, stream>>>(w_out_s, woT_s, D, D);

  // ---- temporal attention (attend over t, per (b,c)) ----
  ln_cast_kernel<<<NTOK / 8, 256, 0, stream>>>(x, gamma_t, beta_t, h, NTOK);
  {
    int waves = (NTOK / 32) * (NI3 / 64);
    gemm_bf16_kernel<<<waves / 8, 256, 0, stream>>>(h, wqT_t, NTOK, NI3, qkv,
                                                    nullptr, nullptr);
  }
  flash_kernel<2, true><<<2 * 1024, 256, 0, stream>>>(qkv, attn);
  {
    int waves = (NTOK / 32) * (D / 64);
    gemm_bf16_kernel<<<waves / 8, 256, 0, stream>>>(attn, woT_t, NTOK, D,
                                                    nullptr, mid, b_out_t);
  }

  // ---- spatial attention (attend over c, per (b,t)) ----
  ln_cast_kernel<<<NTOK / 8, 256, 0, stream>>>(mid, gamma_s, beta_s, h, NTOK);
  {
    int waves = (NTOK / 32) * (NI3 / 64);
    gemm_bf16_kernel<<<waves / 8, 256, 0, stream>>>(h, wqT_s, NTOK, NI3, qkv,
                                                    nullptr, nullptr);
  }
  flash_kernel<1, false><<<2 * 32 * 64, 256, 0, stream>>>(qkv, attn);
  {
    int waves = (NTOK / 32) * (D / 64);
    gemm_bf16_kernel<<<waves / 8, 256, 0, stream>>>(attn, woT_s, NTOK, D,
                                                    nullptr, out, b_out_s);
  }
}
